// SelfAttention_66468913873622
// MI455X (gfx1250) — matrix operand order repaired and checked by reading, not compiled, not run on hardware
//
#include <hip/hip_runtime.h>
#include <hip/hip_bf16.h>

typedef __attribute__((ext_vector_type(16))) _Float16 v16h;
typedef __attribute__((ext_vector_type(8)))  _Float16 v8h;
typedef __attribute__((ext_vector_type(4)))  _Float16 v4h;
typedef __attribute__((ext_vector_type(8)))  float    v8f;

#define D_IN   1024
#define D_OUT  1024
#define SEQ_T  2048
#define BATCH  4
#define NROWS  (BATCH * SEQ_T)   // 8192

union F16x16 {
    v16h v;
    struct { v8h lo; v8h hi; } p;
};

static __device__ __forceinline__ v8f wmma_f16(v16h a, v16h b, v8f c) {
    return __builtin_amdgcn_wmma_f32_16x16x32_f16(
        false, a, false, b, (short)0, c, false, false);
}

// ---------------------------------------------------------------------------
// Kernel 1: convert x (fp32) -> f16, vectorized 4-wide
// ---------------------------------------------------------------------------
__global__ __launch_bounds__(256) void cvt_x(const float* __restrict__ x,
                                             _Float16* __restrict__ xh) {
    int i = blockIdx.x * 256 + threadIdx.x;   // 2097152 vec4 elements
    if (i < (NROWS * D_IN) / 4) {
        float4 f = ((const float4*)x)[i];
        v4h h;
        h.x = (_Float16)f.x; h.y = (_Float16)f.y;
        h.z = (_Float16)f.z; h.w = (_Float16)f.w;
        ((v4h*)xh)[i] = h;
    }
}

// ---------------------------------------------------------------------------
// Kernel 2: convert + transpose weights: Wt[z][n][k] = (f16) W[k][n]
// ---------------------------------------------------------------------------
__global__ __launch_bounds__(256) void cvt_w(const float* __restrict__ Wq,
                                             const float* __restrict__ Wk,
                                             const float* __restrict__ Wv,
                                             _Float16* __restrict__ wt) {
    int z = blockIdx.y;
    const float* W = (z == 0) ? Wq : ((z == 1) ? Wk : Wv);
    _Float16* o = wt + (size_t)z * (D_IN * D_OUT);
    int idx = blockIdx.x * 256 + threadIdx.x;     // 1048576 elements
    int n = idx >> 10;
    int k = idx & 1023;
    o[idx] = (_Float16)W[k * D_OUT + n];
}

// ---------------------------------------------------------------------------
// Kernel 3: QKV projection GEMM. grid = (D_OUT/256, NROWS/32, 3), block 128.
// Each wave: 32x64 output tile, K-loop step 32, 8 wmma per step.
// q,k stored row-major f16; v stored transposed [b][d][t] f16.
// ---------------------------------------------------------------------------
__global__ __launch_bounds__(128) void qkv_gemm(
    const _Float16* __restrict__ xh, const _Float16* __restrict__ wt,
    const float* __restrict__ bq, const float* __restrict__ bk,
    const float* __restrict__ bv,
    _Float16* __restrict__ qh, _Float16* __restrict__ kh,
    _Float16* __restrict__ vT) {

    const int lane = threadIdx.x & 31;
    const int wave = threadIdx.x >> 5;
    const int z    = blockIdx.z;
    const int m0   = blockIdx.y * 32;
    const int nb   = blockIdx.x * 256 + wave * 64;
    const int ln15 = lane & 15;
    const int hf   = lane >> 4;      // 0 for lanes 0-15, 1 for lanes 16-31

    const _Float16* W    = wt + (size_t)z * (D_IN * D_OUT);
    const float*    bias = (z == 0) ? bq : ((z == 1) ? bk : bv);

    v8f acc[2][4];
    #pragma unroll
    for (int mi = 0; mi < 2; ++mi)
        #pragma unroll
        for (int nt = 0; nt < 4; ++nt) {
            float bb = bias[nb + nt * 16 + ln15];
            #pragma unroll
            for (int e = 0; e < 8; ++e) acc[mi][nt][e] = bb;
        }

    for (int kk = 0; kk < D_IN; kk += 32) {
        F16x16 a[2];
        #pragma unroll
        for (int mi = 0; mi < 2; ++mi) {
            const _Float16* p =
                xh + (size_t)(m0 + mi * 16 + ln15) * D_IN + kk + hf * 16;
            a[mi].p.lo = *(const v8h*)p;
            a[mi].p.hi = *(const v8h*)(p + 8);
        }
        #pragma unroll
        for (int nt = 0; nt < 4; ++nt) {
            const _Float16* pb =
                W + (size_t)(nb + nt * 16 + ln15) * D_IN + kk + hf * 16;
            v16h bfr = *(const v16h*)pb;
            #pragma unroll
            for (int mi = 0; mi < 2; ++mi)
                acc[mi][nt] = wmma_f16(a[mi].v, bfr, acc[mi][nt]);
        }
    }

    if (z < 2) {
        _Float16* outp = (z == 0) ? qh : kh;
        #pragma unroll
        for (int mi = 0; mi < 2; ++mi)
            #pragma unroll
            for (int nt = 0; nt < 4; ++nt) {
                int n = nb + nt * 16 + ln15;
                #pragma unroll
                for (int r = 0; r < 8; ++r) {
                    int m = m0 + mi * 16 + r + hf * 8;
                    outp[(size_t)m * D_OUT + n] = (_Float16)acc[mi][nt][r];
                }
            }
    } else {
        // V transposed: vT[b][n][t]; 8 consecutive t values per lane -> one b128
        #pragma unroll
        for (int mi = 0; mi < 2; ++mi) {
            int mbase = m0 + mi * 16 + hf * 8;
            int bidx = mbase >> 11;
            int t0   = mbase & (SEQ_T - 1);
            #pragma unroll
            for (int nt = 0; nt < 4; ++nt) {
                int n = nb + nt * 16 + ln15;
                v8h pack;
                #pragma unroll
                for (int r = 0; r < 8; ++r) pack[r] = (_Float16)acc[mi][nt][r];
                *(v8h*)(vT + (size_t)bidx * D_OUT * SEQ_T +
                        (size_t)n * SEQ_T + t0) = pack;
            }
        }
    }
}

// ---------------------------------------------------------------------------
// One 32-key tile of flash attention. DIAG=false: fully unmasked, branchless
// (the hot path for kt < blockIdx.x). DIAG=true: causal tile, masked with
// selects (instantiated exactly once per block).
// ---------------------------------------------------------------------------
template <bool DIAG>
static __device__ __forceinline__ void attn_tile(
    int kt, int lastkt, int tid, int lane, int mi, int ln15, int hf,
    int row, int q0, int dbase, size_t rowbase,
    const _Float16* __restrict__ qh, const _Float16* __restrict__ kh,
    const _Float16* __restrict__ vTb,
    float (&Sbuf)[32][33], float& mreg, float& lreg, v8f (&o)[16]) {

    const float scale = 0.03125f;              // 1/sqrt(1024)

    // zero the score reduction buffer
    for (int i = tid; i < 32 * 33; i += 256) ((float*)Sbuf)[i] = 0.0f;
    __syncthreads();

    // hint-prefetch next key tile's K and V (global_prefetch_b8)
    if (!DIAG) {
        __builtin_prefetch(
            kh + (rowbase + (kt + 1) * 32 + lane) * D_IN + dbase, 0, 1);
        __builtin_prefetch(
            vTb + (size_t)(dbase + lane * 8) * SEQ_T + (kt + 1) * 32, 0, 1);
    }

    // partial S over this wave's 256-dim slice: 16 wmma
    v8f s[2];
    #pragma unroll
    for (int nt = 0; nt < 2; ++nt)
        #pragma unroll
        for (int e = 0; e < 8; ++e) s[nt][e] = 0.0f;

    #pragma unroll
    for (int ks = 0; ks < 8; ++ks) {
        int d = dbase + ks * 32;
        F16x16 a;
        const _Float16* p =
            qh + (rowbase + q0 + mi * 16 + ln15) * D_IN + d + hf * 16;
        a.p.lo = *(const v8h*)p;
        a.p.hi = *(const v8h*)(p + 8);
        #pragma unroll
        for (int nt = 0; nt < 2; ++nt) {
            const _Float16* pk =
                kh + (rowbase + kt * 32 + nt * 16 + ln15) * D_IN + d + hf * 16;
            v16h bfr = *(const v16h*)pk;
            s[nt] = wmma_f16(a.v, bfr, s[nt]);
        }
    }

    // cross-slice reduction into LDS (ds_add_f32)
    #pragma unroll
    for (int nt = 0; nt < 2; ++nt)
        #pragma unroll
        for (int r = 0; r < 8; ++r)
            atomicAdd(&Sbuf[mi * 16 + r + hf * 8][nt * 16 + ln15], s[nt][r]);
    __syncthreads();

    // ---- parallel online softmax: each lane handles its own row ----
    float mold = mreg;
    float mx = mold;
    #pragma unroll
    for (int j = 0; j < 32; ++j) {
        float sj = Sbuf[row][j] * scale;
        if (DIAG) sj = (j <= row) ? sj : -__builtin_inff();
        mx = fmaxf(mx, sj);
    }
    float c = __expf(mold - mx);
    mreg = mx;
    float lacc = lreg * c;

    // build P A-fragment in registers: this lane needs cols
    // [hf*16, hf*16+16); the other 16 cols only
    // contribute to the row sum.
    F16x16 pa;
    const int c0 = hf * 16;
    const int c1 = (hf ^ 1) * 16;
    #pragma unroll
    for (int jj = 0; jj < 8; ++jj) {
        int j = c0 + jj;
        float pj = __expf(Sbuf[row][j] * scale - mx);
        if (DIAG) pj = (j <= row) ? pj : 0.0f;
        pa.p.lo[jj] = (_Float16)pj;
        lacc += pj;
    }
    #pragma unroll
    for (int jj = 0; jj < 8; ++jj) {
        int j = 8 + c0 + jj;
        float pj = __expf(Sbuf[row][j] * scale - mx);
        if (DIAG) pj = (j <= row) ? pj : 0.0f;
        pa.p.hi[jj] = (_Float16)pj;
        lacc += pj;
    }
    #pragma unroll
    for (int jj = 0; jj < 8; ++jj) {
        int j = c1 + jj;
        float pj = __expf(Sbuf[row][j] * scale - mx);
        if (DIAG) pj = (j <= row) ? pj : 0.0f;
        lacc += pj;
    }
    #pragma unroll
    for (int jj = 0; jj < 8; ++jj) {
        int j = 8 + c1 + jj;
        float pj = __expf(Sbuf[row][j] * scale - mx);
        if (DIAG) pj = (j <= row) ? pj : 0.0f;
        lacc += pj;
    }
    lreg = lacc;

    // per-row rescale factors: rows 0..15 live in lanes 0..15
    float cr[8];
    #pragma unroll
    for (int r = 0; r < 8; ++r) cr[r] = __shfl(c, r + hf * 8, 32);

    // rescale accumulators + PV (16 wmma)
    #pragma unroll
    for (int nt = 0; nt < 16; ++nt) {
        #pragma unroll
        for (int r = 0; r < 8; ++r) o[nt][r] *= cr[r];
        const _Float16* pv =
            vTb + (size_t)(dbase + nt * 16 + ln15) * SEQ_T + kt * 32 + hf * 16;
        v16h bfr = *(const v16h*)pv;
        o[nt] = wmma_f16(pa.v, bfr, o[nt]);
    }
    __syncthreads();   // protect Sbuf reads before next tile's zeroing
}

// ---------------------------------------------------------------------------
// Kernel 4: flash attention. grid = (T/32, B), block 256 (8 waves).
// wave = (mi, dj): mi in {0,1} -> 16-row subtile, dj in {0..3} -> 256-dim slice.
// Softmax is computed redundantly by every wave for its own 16 rows
// (lane ln15 owns row mi*16+ln15); P fragments are built in registers,
// per-row rescale factors distributed with wave shuffles. The causal
// (diagonal) tile is peeled out of the hot loop.
// ---------------------------------------------------------------------------
__global__ __launch_bounds__(256) void attn(
    const _Float16* __restrict__ qh, const _Float16* __restrict__ kh,
    const _Float16* __restrict__ vT, float* __restrict__ out) {

    const int tid  = threadIdx.x;
    const int lane = tid & 31;
    const int wave = tid >> 5;
    const int mi   = wave & 1;
    const int dj   = wave >> 1;
    const int ln15 = lane & 15;
    const int hf   = lane >> 4;
    const int b    = blockIdx.y;
    const int q0   = blockIdx.x * 32;
    const int dbase = dj * 256;

    __shared__ float Sbuf[32][33];

    v8f o[16];
    #pragma unroll
    for (int nt = 0; nt < 16; ++nt)
        #pragma unroll
        for (int e = 0; e < 8; ++e) o[nt][e] = 0.0f;

    // per-lane running softmax state for query row (mi*16 + ln15)
    const int row = mi * 16 + ln15;
    float mreg = -__builtin_inff();
    float lreg = 0.0f;

    const size_t rowbase = (size_t)b * SEQ_T;
    const _Float16* vTb  = vT + (size_t)b * D_OUT * SEQ_T;

    const int lastkt = (int)blockIdx.x;
    for (int kt = 0; kt < lastkt; ++kt)
        attn_tile<false>(kt, lastkt, tid, lane, mi, ln15, hf, row, q0, dbase,
                         rowbase, qh, kh, vTb, Sbuf, mreg, lreg, o);
    attn_tile<true>(lastkt, lastkt, tid, lane, mi, ln15, hf, row, q0, dbase,
                    rowbase, qh, kh, vTb, Sbuf, mreg, lreg, o);

    // normalize and write fp32 output
    float invl[8];
    #pragma unroll
    for (int r = 0; r < 8; ++r)
        invl[r] = 1.0f / __shfl(lreg, r + hf * 8, 32);

    #pragma unroll
    for (int nt = 0; nt < 16; ++nt) {
        int d = dbase + nt * 16 + ln15;
        #pragma unroll
        for (int r = 0; r < 8; ++r) {
            int m = q0 + mi * 16 + r + hf * 8;
            out[(rowbase + m) * (size_t)D_OUT + d] = o[nt][r] * invl[r];
        }
    }
}

// ---------------------------------------------------------------------------
// Workspace layout (bytes):
//   xh : 0         .. 16777216   (8192*1024 f16)
//   wt : 16777216  .. 23068672   (3 * 1024*1024 f16, transposed)
//   qh : 23068672  .. 39845888
//   kh : 39845888  .. 56623104
//   vT : 56623104  .. 73400320   ([b][d][t] f16)
// ---------------------------------------------------------------------------
extern "C" void kernel_launch(void* const* d_in, const int* in_sizes, int n_in,
                              void* d_out, int out_size, void* d_ws,
                              size_t ws_size, hipStream_t stream) {
    const float* x  = (const float*)d_in[0];
    const float* Wq = (const float*)d_in[1];
    const float* bq = (const float*)d_in[2];
    const float* Wk = (const float*)d_in[3];
    const float* bk = (const float*)d_in[4];
    const float* Wv = (const float*)d_in[5];
    const float* bv = (const float*)d_in[6];
    float* out = (float*)d_out;

    char* ws = (char*)d_ws;
    _Float16* xh = (_Float16*)(ws + 0);
    _Float16* wt = (_Float16*)(ws + 16777216);
    _Float16* qh = (_Float16*)(ws + 23068672);
    _Float16* kh = (_Float16*)(ws + 39845888);
    _Float16* vT = (_Float16*)(ws + 56623104);

    cvt_x<<<dim3((NROWS * D_IN / 4 + 255) / 256), 256, 0, stream>>>(x, xh);
    cvt_w<<<dim3(4096, 3), 256, 0, stream>>>(Wq, Wk, Wv, wt);
    qkv_gemm<<<dim3(D_OUT / 256, NROWS / 32, 3), 128, 0, stream>>>(
        xh, wt, bq, bk, bv, qh, kh, vT);
    attn<<<dim3(SEQ_T / 32, BATCH), 256, 0, stream>>>(qh, kh, vT, out);
}
